// Attention4D_53523882443536
// MI455X (gfx1250) — matrix operand order repaired and checked by reading, not compiled, not run on hardware
//
#include <hip/hip_runtime.h>
#include <hip/hip_bf16.h>
#include <math.h>

// ---------------- problem constants ----------------
#define BB     128
#define DIMC   384
#define RESHW  14
#define NTOK   196          // RES*RES
#define NP     224          // padded token count (14 tiles of 16)
#define NHEAD  8
#define KD     32           // key dim (== one WMMA K step!)
#define DV     128          // value dim per head
#define DHTOT  1024         // NHEAD*DV
#define OQKV   1536         // 256 q + 256 k + 1024 v output channels
#define QSCALE 0.17677669529663687f  // 32^-0.5

typedef _Float16 h16v __attribute__((ext_vector_type(16)));
typedef float    f8v  __attribute__((ext_vector_type(8)));

// ---------------- WMMA helpers ----------------
__device__ __forceinline__ f8v fzero8() {
    f8v z;
#pragma unroll
    for (int i = 0; i < 8; ++i) z[i] = 0.0f;
    return z;
}

__device__ __forceinline__ f8v wmma_f16(h16v a, h16v b, f8v c) {
    // D = A(16x32 f16) * B(32x16 f16) + C(16x16 f32)
    return __builtin_amdgcn_wmma_f32_16x16x32_f16(
        false, a, false, b, (short)0, c, false, false);
}

// A fragment: 16x32, row = lane&15, K split {kb..kb+7, kb+16..kb+23}, kb=(lane>>4)*8
__device__ __forceinline__ h16v load_frag_a(const _Float16* p0, int ld) {
    int lane = threadIdx.x & 31;
    const _Float16* p = p0 + (size_t)(lane & 15) * ld + (lane >> 4) * 8;
    h16v a;
#pragma unroll
    for (int e = 0; e < 8; ++e) { a[e] = p[e]; a[e + 8] = p[e + 16]; }
    return a;
}

// B fragment: 32x16, col = lane&15 (stored row-major [col][K]), K split {kb..kb+7, kb+16..kb+23}, kb=(lane>>4)*8
__device__ __forceinline__ h16v load_frag_b(const _Float16* p0, int ld) {
    int lane = threadIdx.x & 31;
    const _Float16* p = p0 + (size_t)(lane & 15) * ld + (lane >> 4) * 8;
    h16v b;
#pragma unroll
    for (int e = 0; e < 8; ++e) { b[e] = p[e]; b[e + 8] = p[e + 16]; }
    return b;
}

// ---------------- pack kernels (fold BN into weights) ----------------
__global__ void pack_qkvw_kernel(const float* __restrict__ qw, const float* __restrict__ qb,
                                 const float* __restrict__ qs, const float* __restrict__ qt,
                                 const float* __restrict__ kw, const float* __restrict__ kb,
                                 const float* __restrict__ ks, const float* __restrict__ kt,
                                 const float* __restrict__ vw, const float* __restrict__ vb,
                                 const float* __restrict__ vs, const float* __restrict__ vt,
                                 _Float16* __restrict__ W, float* __restrict__ bias) {
    int i = blockIdx.x * 256 + threadIdx.x;
    if (i >= OQKV * DIMC) return;
    int c = i % DIMC, o = i / DIMC;
    float w, s;
    if (o < 256)      { w = qw[o * DIMC + c];        s = qs[o]; }
    else if (o < 512) { int o2 = o - 256; w = kw[o2 * DIMC + c]; s = ks[o2]; }
    else              { int o2 = o - 512; w = vw[o2 * DIMC + c]; s = vs[o2]; }
    W[i] = (_Float16)(w * s);
    if (c == 0) {
        if (o < 256)      bias[o] = qs[o] * qb[o] + qt[o];
        else if (o < 512) { int o2 = o - 256; bias[o] = ks[o2] * kb[o2] + kt[o2]; }
        else              { int o2 = o - 512; bias[o] = vs[o2] * vb[o2] + vt[o2]; }
    }
}

__global__ void pack_projw_kernel(const float* __restrict__ pw, const float* __restrict__ pb,
                                  const float* __restrict__ ps, const float* __restrict__ pt,
                                  _Float16* __restrict__ Wp, float* __restrict__ bp) {
    int i = blockIdx.x * 256 + threadIdx.x;
    if (i >= DIMC * DHTOT) return;
    int o = i / DHTOT;
    Wp[i] = (_Float16)(ps[o] * pw[i]);
    if ((i % DHTOT) == 0) bp[o] = ps[o] * pb[o] + pt[o];
}

__global__ void pack_vlw_kernel(const float* __restrict__ vlw, const float* __restrict__ vlb,
                                const float* __restrict__ vls, const float* __restrict__ vlt,
                                float* __restrict__ wl, float* __restrict__ bl) {
    int i = blockIdx.x * 256 + threadIdx.x;
    if (i < DHTOT * 9) { int c = i / 9; wl[i] = vls[c] * vlw[i]; }
    if (i < DHTOT)     bl[i] = vls[i] * vlb[i] + vlt[i];
}

// x (B,DIM,N) f32 -> xh (B,NP,DIM) f16, padded rows zero
__global__ void pack_x_kernel(const float* __restrict__ x, _Float16* __restrict__ xh) {
    size_t i = (size_t)blockIdx.x * 256 + threadIdx.x;
    if (i >= (size_t)BB * NP * DIMC) return;
    int c = (int)(i % DIMC);
    size_t t = i / DIMC;
    int n = (int)(t % NP);
    int b = (int)(t / NP);
    float v = (n < NTOK) ? x[((size_t)b * DIMC + c) * NTOK + n] : 0.0f;
    xh[i] = (_Float16)v;
}

// ---------------- QKV GEMM: (1536x384) @ (384x224) per batch ----------------
// Register-blocked: each wave computes a 64x32 macro-tile = 4x2 WMMA tiles.
// Per K-step: 4 A-frag + 2 B-frag loads feed 8 WMMAs (A reused 2x, B reused 4x).
__global__ __launch_bounds__(256) void qkv_gemm_kernel(
    const _Float16* __restrict__ W, const float* __restrict__ bias,
    const _Float16* __restrict__ xh,
    _Float16* __restrict__ qh, _Float16* __restrict__ kh, _Float16* __restrict__ vh) {
    int wid  = threadIdx.x >> 5;
    size_t t = (size_t)blockIdx.x * 8 + wid;
    if (t >= (size_t)BB * 24 * 7) return;          // 24 o-groups(64), 7 n-groups(32)
    int b  = (int)(t / (24 * 7));
    int rr = (int)(t % (24 * 7));
    int og = rr / 7, ng = rr % 7;
    int lane = threadIdx.x & 31, lc = lane & 15, hi = lane >> 4;

    const _Float16* arow = W  + (size_t)(og * 64) * DIMC;
    const _Float16* brow = xh + ((size_t)b * NP + ng * 32) * DIMC;

    f8v acc[4][2];
#pragma unroll
    for (int i = 0; i < 4; ++i)
#pragma unroll
        for (int j = 0; j < 2; ++j) acc[i][j] = fzero8();

#pragma unroll 2
    for (int k0 = 0; k0 < DIMC; k0 += 32) {
        h16v af[4], bf[2];
#pragma unroll
        for (int i = 0; i < 4; ++i)
            af[i] = load_frag_a(arow + (size_t)(i * 16) * DIMC + k0, DIMC);
#pragma unroll
        for (int j = 0; j < 2; ++j)
            bf[j] = load_frag_b(brow + (size_t)(j * 16) * DIMC + k0, DIMC);
#pragma unroll
        for (int i = 0; i < 4; ++i)
#pragma unroll
            for (int j = 0; j < 2; ++j)
                acc[i][j] = wmma_f16(af[i], bf[j], acc[i][j]);
    }

#pragma unroll
    for (int j = 0; j < 2; ++j) {
        int n = ng * 32 + j * 16 + lc;
#pragma unroll
        for (int i = 0; i < 4; ++i) {
#pragma unroll
            for (int r = 0; r < 8; ++r) {
                int m = og * 64 + i * 16 + r + 8 * hi;
                float val = acc[i][j][r] + bias[m];
                if (m < 256) {          // q, pre-scaled by 1/sqrt(KD)
                    int h = m >> 5, kd = m & 31;
                    qh[(((size_t)b * NHEAD + h) * NP + n) * KD + kd] =
                        (_Float16)(val * QSCALE);
                } else if (m < 512) {   // k
                    int m2 = m - 256; int h = m2 >> 5, kd = m2 & 31;
                    kh[(((size_t)b * NHEAD + h) * NP + n) * KD + kd] = (_Float16)val;
                } else {                // v (zero padded tokens so attn pads are inert)
                    int c = m - 512;
                    vh[((size_t)b * DHTOT + c) * NP + n] =
                        (_Float16)(n < NTOK ? val : 0.0f);
                }
            }
        }
    }
}

// ---------------- depthwise 3x3 local-V (BN folded) ----------------
__global__ void vloc_kernel(const _Float16* __restrict__ vh,
                            const float* __restrict__ wl, const float* __restrict__ bl,
                            _Float16* __restrict__ vloc) {
    size_t i = (size_t)blockIdx.x * 256 + threadIdx.x;
    if (i >= (size_t)BB * DHTOT * NTOK) return;
    int n = (int)(i % NTOK);
    size_t t = i / NTOK;
    int c = (int)(t % DHTOT);
    int b = (int)(t / DHTOT);
    int y = n / RESHW, x = n % RESHW;
    float acc = bl[c];
    const _Float16* base = vh + ((size_t)b * DHTOT + c) * NP;
#pragma unroll
    for (int ky = 0; ky < 3; ++ky) {
        int yy = y + ky - 1;
        if (yy < 0 || yy >= RESHW) continue;
#pragma unroll
        for (int kx = 0; kx < 3; ++kx) {
            int xx = x + kx - 1;
            if (xx < 0 || xx >= RESHW) continue;
            acc += wl[c * 9 + ky * 3 + kx] * (float)base[yy * RESHW + xx];
        }
    }
    vloc[((size_t)b * DHTOT + c) * NP + n] = (_Float16)acc;
}

// ---------------- fused attention: QK^T + bias + th1 + softmax + th2 + attn@V + vloc + relu ----
__global__ __launch_bounds__(256) void attn_kernel(
    const _Float16* __restrict__ qh, const _Float16* __restrict__ kh,
    const _Float16* __restrict__ vh, const _Float16* __restrict__ vloc,
    const float* __restrict__ th1_w, const float* __restrict__ th1_b,
    const float* __restrict__ th2_w, const float* __restrict__ th2_b,
    const float* __restrict__ bias_seg, const int* __restrict__ bias_idxs,
    int n_off, _Float16* __restrict__ xout) {
    __shared__ _Float16 S[NHEAD][16][NP];   // scores -> probs (in-place after barrier)
    __shared__ _Float16 A2[NHEAD][16][NP];  // post-th2 attention (f16 A-fragments)

    int b  = blockIdx.x / 13;
    int nt = blockIdx.x % 13;
    int n0 = nt * 16;
    int o    = threadIdx.x >> 5;   // wave == head
    int lane = threadIdx.x & 31, lc = lane & 15, hi = lane >> 4;

    float t1[NHEAD], t2[NHEAD];
#pragma unroll
    for (int i = 0; i < NHEAD; ++i) {
        t1[i] = th1_w[o * NHEAD + i];
        t2[i] = th2_w[o * NHEAD + i];
    }
    float b1 = th1_b[o], b2 = th2_b[o];

    // ---- phase 1: raw scores (one WMMA per 16x16 tile; K = KD = 32) + bias ----
    h16v aq = load_frag_a(qh + (((size_t)b * NHEAD + o) * NP + n0) * KD, KD);
#pragma unroll 1
    for (int mt = 0; mt < 14; ++mt) {
        h16v bk = load_frag_b(kh + (((size_t)b * NHEAD + o) * NP + mt * 16) * KD, KD);
        f8v acc = fzero8();
        acc = wmma_f16(aq, bk, acc);
        int m = mt * 16 + lc;
#pragma unroll
        for (int r = 0; r < 8; ++r) {
            int nr = r + 8 * hi;
            int ng = n0 + nr;
            float s;
            if (m < NTOK) {
                s = acc[r];
                if (ng < NTOK)
                    s += bias_seg[(size_t)o * n_off + bias_idxs[ng * NTOK + m]];
            } else {
                s = -65504.0f;      // padded key -> masked out of softmax
            }
            S[o][nr][m] = (_Float16)s;
        }
    }
    __syncthreads();

    // ---- phase 2: th1 head-mix + softmax (rows in 2 chunks of 8; regs + wave32 shfl) ----
#pragma unroll 1
    for (int ch = 0; ch < 2; ++ch) {
        float mx[8][7];
#pragma unroll
        for (int rr = 0; rr < 8; ++rr) {
            int nr = ch * 8 + rr;
#pragma unroll
            for (int j = 0; j < 7; ++j) {
                int m = lane + 32 * j;
                float a = b1;
#pragma unroll
                for (int i = 0; i < NHEAD; ++i) a += t1[i] * (float)S[i][nr][m];
                mx[rr][j] = a;
            }
        }
        __syncthreads();   // all waves done reading S rows of this chunk
#pragma unroll
        for (int rr = 0; rr < 8; ++rr) {
            int nr = ch * 8 + rr;
            float rmax = mx[rr][0];
#pragma unroll
            for (int j = 1; j < 7; ++j) rmax = fmaxf(rmax, mx[rr][j]);
#pragma unroll
            for (int off = 16; off >= 1; off >>= 1)
                rmax = fmaxf(rmax, __shfl_xor(rmax, off, 32));
            float e[7], sum = 0.0f;
#pragma unroll
            for (int j = 0; j < 7; ++j) { e[j] = __expf(mx[rr][j] - rmax); sum += e[j]; }
#pragma unroll
            for (int off = 16; off >= 1; off >>= 1)
                sum += __shfl_xor(sum, off, 32);
            float inv = 1.0f / sum;
#pragma unroll
            for (int j = 0; j < 7; ++j)
                S[o][nr][lane + 32 * j] = (_Float16)(e[j] * inv);
        }
        __syncthreads();
    }

    // ---- phase 2b: th2 head-mix -> f16 attention matrix ----
#pragma unroll 1
    for (int nr = 0; nr < 16; ++nr) {
#pragma unroll
        for (int j = 0; j < 7; ++j) {
            int m = lane + 32 * j;
            float a = b2;
#pragma unroll
            for (int i = 0; i < NHEAD; ++i) a += t2[i] * (float)S[i][nr][m];
            if (m >= NTOK) a = 0.0f;   // pads contribute nothing (v pads are zero too)
            A2[o][nr][m] = (_Float16)a;
        }
    }
    __syncthreads();

    // ---- phase 3: out = attn @ V  (8 d-tiles x 7 K-steps of 32) + vloc + relu ----
    f8v accD[8];
#pragma unroll
    for (int dt = 0; dt < 8; ++dt) accD[dt] = fzero8();
#pragma unroll 1
    for (int kt = 0; kt < 7; ++kt) {
        h16v a = load_frag_a(&A2[o][0][0] + kt * 32, NP);
#pragma unroll
        for (int dt = 0; dt < 8; ++dt) {
            h16v bv = load_frag_b(
                vh + ((size_t)b * DHTOT + o * DV + dt * 16) * NP + kt * 32, NP);
            accD[dt] = wmma_f16(a, bv, accD[dt]);
        }
    }
#pragma unroll
    for (int dt = 0; dt < 8; ++dt) {
        int d = dt * 16 + lc;
        int c = o * DV + d;
#pragma unroll
        for (int r = 0; r < 8; ++r) {
            int ng = n0 + r + 8 * hi;
            float v = accD[dt][r] + (float)vloc[((size_t)b * DHTOT + c) * NP + ng];
            v = v > 0.0f ? v : 0.0f;
            xout[((size_t)b * NP + ng) * DHTOT + c] = (_Float16)v;
        }
    }
}

// ---------------- projection GEMM: (384x1024) @ (1024x224) per batch ----------------
// Register-blocked 4x2 like the QKV GEMM. n padded to 224; stores masked to n<196.
__global__ __launch_bounds__(256) void proj_gemm_kernel(
    const _Float16* __restrict__ Wp, const float* __restrict__ bp,
    const _Float16* __restrict__ xout, float* __restrict__ out) {
    int wid  = threadIdx.x >> 5;
    size_t t = (size_t)blockIdx.x * 8 + wid;
    if (t >= (size_t)BB * 6 * 7) return;           // 6 c-groups(64), 7 n-groups(32)
    int b  = (int)(t / (6 * 7));
    int rr = (int)(t % (6 * 7));
    int cg = rr / 7, ng = rr % 7;
    int lane = threadIdx.x & 31, lc = lane & 15, hi = lane >> 4;

    const _Float16* arow = Wp   + (size_t)(cg * 64) * DHTOT;
    const _Float16* brow = xout + ((size_t)b * NP + ng * 32) * DHTOT;

    f8v acc[4][2];
#pragma unroll
    for (int i = 0; i < 4; ++i)
#pragma unroll
        for (int j = 0; j < 2; ++j) acc[i][j] = fzero8();

#pragma unroll 2
    for (int k0 = 0; k0 < DHTOT; k0 += 32) {
        h16v af[4], bf[2];
#pragma unroll
        for (int i = 0; i < 4; ++i)
            af[i] = load_frag_a(arow + (size_t)(i * 16) * DHTOT + k0, DHTOT);
#pragma unroll
        for (int j = 0; j < 2; ++j)
            bf[j] = load_frag_b(brow + (size_t)(j * 16) * DHTOT + k0, DHTOT);
#pragma unroll
        for (int i = 0; i < 4; ++i)
#pragma unroll
            for (int j = 0; j < 2; ++j)
                acc[i][j] = wmma_f16(af[i], bf[j], acc[i][j]);
    }

#pragma unroll
    for (int j = 0; j < 2; ++j) {
        int n = ng * 32 + j * 16 + lc;
        if (n >= NTOK) continue;
#pragma unroll
        for (int i = 0; i < 4; ++i) {
#pragma unroll
            for (int r = 0; r < 8; ++r) {
                int co = cg * 64 + i * 16 + r + 8 * hi;
                out[((size_t)b * DIMC + co) * NTOK + n] = acc[i][j][r] + bp[co];
            }
        }
    }
}

// ---------------- launch ----------------
extern "C" void kernel_launch(void* const* d_in, const int* in_sizes, int n_in,
                              void* d_out, int out_size, void* d_ws, size_t ws_size,
                              hipStream_t stream) {
    const float* x        = (const float*)d_in[0];
    const float* q_w      = (const float*)d_in[1];
    const float* q_b      = (const float*)d_in[2];
    const float* q_scale  = (const float*)d_in[3];
    const float* q_shift  = (const float*)d_in[4];
    const float* k_w      = (const float*)d_in[5];
    const float* k_b      = (const float*)d_in[6];
    const float* k_scale  = (const float*)d_in[7];
    const float* k_shift  = (const float*)d_in[8];
    const float* v_w      = (const float*)d_in[9];
    const float* v_b      = (const float*)d_in[10];
    const float* v_scale  = (const float*)d_in[11];
    const float* v_shift  = (const float*)d_in[12];
    const float* vl_w     = (const float*)d_in[13];
    const float* vl_b     = (const float*)d_in[14];
    const float* vl_scale = (const float*)d_in[15];
    const float* vl_shift = (const float*)d_in[16];
    const float* th1_w    = (const float*)d_in[17];
    const float* th1_b    = (const float*)d_in[18];
    const float* th2_w    = (const float*)d_in[19];
    const float* th2_b    = (const float*)d_in[20];
    const float* proj_w   = (const float*)d_in[21];
    const float* proj_b   = (const float*)d_in[22];
    const float* proj_sc  = (const float*)d_in[23];
    const float* proj_sh  = (const float*)d_in[24];
    const float* bias_seg = (const float*)d_in[25];
    const int*   bias_idx = (const int*)d_in[26];
    int n_off = in_sizes[25] / NHEAD;

    uint8_t* w = (uint8_t*)d_ws;
    size_t off = 0;
    auto alloc = [&](size_t bytes) -> void* {
        void* p = w + off;
        off = (off + bytes + 255) & ~(size_t)255;
        return p;
    };
    _Float16* Wqkv  = (_Float16*)alloc((size_t)OQKV * DIMC * 2);
    float*    bqkv  = (float*)   alloc((size_t)OQKV * 4);
    _Float16* Wp    = (_Float16*)alloc((size_t)DIMC * DHTOT * 2);
    float*    bp    = (float*)   alloc((size_t)DIMC * 4);
    float*    wl    = (float*)   alloc((size_t)DHTOT * 9 * 4);
    float*    bl    = (float*)   alloc((size_t)DHTOT * 4);
    _Float16* xh    = (_Float16*)alloc((size_t)BB * NP * DIMC * 2);
    _Float16* qh    = (_Float16*)alloc((size_t)BB * NHEAD * NP * KD * 2);
    _Float16* kh    = (_Float16*)alloc((size_t)BB * NHEAD * NP * KD * 2);
    _Float16* vh    = (_Float16*)alloc((size_t)BB * DHTOT * NP * 2);
    _Float16* vloc  = (_Float16*)alloc((size_t)BB * DHTOT * NP * 2);
    _Float16* xout  = (_Float16*)alloc((size_t)BB * NP * DHTOT * 2);
    (void)ws_size;

    pack_qkvw_kernel<<<(OQKV * DIMC + 255) / 256, 256, 0, stream>>>(
        q_w, q_b, q_scale, q_shift, k_w, k_b, k_scale, k_shift,
        v_w, v_b, v_scale, v_shift, Wqkv, bqkv);
    pack_projw_kernel<<<(DIMC * DHTOT + 255) / 256, 256, 0, stream>>>(
        proj_w, proj_b, proj_sc, proj_sh, Wp, bp);
    pack_vlw_kernel<<<(DHTOT * 9 + 255) / 256, 256, 0, stream>>>(
        vl_w, vl_b, vl_scale, vl_shift, wl, bl);
    pack_x_kernel<<<(int)(((size_t)BB * NP * DIMC + 255) / 256), 256, 0, stream>>>(x, xh);

    qkv_gemm_kernel<<<(BB * 24 * 7) / 8, 256, 0, stream>>>(Wqkv, bqkv, xh, qh, kh, vh);

    vloc_kernel<<<(int)(((size_t)BB * DHTOT * NTOK + 255) / 256), 256, 0, stream>>>(
        vh, wl, bl, vloc);

    attn_kernel<<<BB * 13, 256, 0, stream>>>(
        qh, kh, vh, vloc, th1_w, th1_b, th2_w, th2_b,
        bias_seg, bias_idx, n_off, xout);

    proj_gemm_kernel<<<(BB * 6 * 7) / 8, 256, 0, stream>>>(
        Wp, bp, xout, (float*)d_out);
}
